// GraphNet_80573586473054
// MI455X (gfx1250) — compile-verified, matrix operand order checked by reading, not run on hardware
//
#include <hip/hip_runtime.h>
#include <hip/hip_bf16.h>
#include <math.h>

// ---------------- problem constants ----------------
#define N_REG 50000
#define N_HYP 1000
#define N_TOT 51000
#define NE    600000
#define D     128

typedef __attribute__((ext_vector_type(16))) _Float16 v16h;
typedef __attribute__((ext_vector_type(8)))  _Float16 v8h;
typedef __attribute__((ext_vector_type(4)))  _Float16 v4h;
typedef __attribute__((ext_vector_type(8)))  float    v8f;

// ---------------- workspace layout (bytes) ----------------
#define WS_EW0P  ((size_t)0)                       // 384*128 halves
#define WS_EW1P  (WS_EW0P + 384*128*2)             // 128*128
#define WS_EW2P  (WS_EW1P + 128*128*2)
#define WS_NW0P  (WS_EW2P + 128*128*2)             // 640*128
#define WS_NW1P  (WS_NW0P + 640*128*2)
#define WS_NW2P  (WS_NW1P + 128*128*2)
#define WS_SUM   (WS_NW2P + 128*128*2)             // N_TOT*D f32
#define WS_CNT   (WS_SUM + (size_t)N_TOT*D*4)      // N_TOT f32
#define WS_MAXE  (WS_CNT + (size_t)N_TOT*4)        // N_TOT*D u32 (encoded)
#define WS_MINE  (WS_MAXE + (size_t)N_TOT*D*4)     // N_TOT*D u32 (encoded)
#define WS_NEED  (WS_MINE + (size_t)N_TOT*D*4)

// ---------------- helpers ----------------
__device__ __forceinline__ unsigned fenc(float x) {
    unsigned u = __float_as_uint(x);
    return (u & 0x80000000u) ? ~u : (u | 0x80000000u);
}
__device__ __forceinline__ float fdec(unsigned v) {
    unsigned u = (v & 0x80000000u) ? (v & 0x7FFFFFFFu) : ~v;
    return __uint_as_float(u);
}

__device__ __forceinline__ v8f wmma16(v16h a, v16h b, v8f c) {
    return __builtin_amdgcn_wmma_f32_16x16x32_f16(false, a, false, b,
                                                  (short)0, c, false, false);
}

// A fragment: lane<16 -> row m0+lane, K {k0..k0+7, k0+16..k0+23}
//             lane>=16 -> row m0+lane-16, K shifted by +8
__device__ __forceinline__ v16h load_a(const _Float16* base, int strideH,
                                       int m0, int k0, int lane) {
    int row = m0 + (lane & 15);
    int kk  = k0 + ((lane & 16) ? 8 : 0);
    const _Float16* p = base + row * strideH + kk;
    v8h lo = *(const v8h*)p;
    v8h hi = *(const v8h*)(p + 16);
    v16h r;
#pragma unroll
    for (int i = 0; i < 8; ++i) { r[i] = lo[i]; r[i + 8] = hi[i]; }
    return r;
}

template <int KT>
__device__ __forceinline__ void gemm8(const _Float16* aLds, int aStride, int m0,
                                      const _Float16* bPack, int lane, v8f* acc) {
    for (int kt = 0; kt < KT; ++kt) {
        v16h a = load_a(aLds, aStride, m0, kt * 32, lane);
#pragma unroll
        for (int n = 0; n < 8; ++n) {
            v16h b = *(const v16h*)(bPack + ((size_t)(kt * 8 + n) * 32 + lane) * 16);
            acc[n] = wmma16(a, b, acc[n]);
        }
    }
}

__device__ __forceinline__ void store_relu_f16(_Float16* hb, int strideH, int m0,
                                               int lane, const v8f* acc,
                                               const float* bias) {
    int col0 = lane & 15;
    int rb   = m0 + ((lane & 16) ? 8 : 0);
#pragma unroll
    for (int n = 0; n < 8; ++n) {
        float bv = bias[n * 16 + col0];
        int col = n * 16 + col0;
#pragma unroll
        for (int i = 0; i < 8; ++i) {
            float v = acc[n][i] + bv;
            v = v > 0.f ? v : 0.f;
            hb[(rb + i) * strideH + col] = (_Float16)v;
        }
    }
}

__device__ __forceinline__ void store_f32(float* h2, int strideF, int m0, int lane,
                                          const v8f* acc, const float* bias) {
    int col0 = lane & 15;
    int rb   = m0 + ((lane & 16) ? 8 : 0);
#pragma unroll
    for (int n = 0; n < 8; ++n) {
        float bv = bias[n * 16 + col0];
        int col = n * 16 + col0;
#pragma unroll
        for (int i = 0; i < 8; ++i)
            h2[(rb + i) * strideF + col] = acc[n][i] + bv;
    }
}

// ---------------- setup kernels ----------------
__global__ void init_ws_kernel(float* sum, float* cnt, unsigned* maxe, unsigned* mine) {
    size_t total = (size_t)N_TOT * D;
    for (size_t i = (size_t)blockIdx.x * blockDim.x + threadIdx.x; i < total;
         i += (size_t)gridDim.x * blockDim.x) {
        sum[i]  = 0.f;
        maxe[i] = 0u;
        mine[i] = 0xFFFFFFFFu;
        if (i < N_TOT) cnt[i] = 0.f;
    }
}

// repack W[K][N] fp32 -> P[(kt*N/16+nt)*32+lane][16] f16 (WMMA B-fragment order)
__global__ void pack_weights_kernel(const float* __restrict__ W,
                                    _Float16* __restrict__ P, int K, int N) {
    int idx = blockIdx.x * 256 + threadIdx.x;
    if (idx >= K * N) return;
    int e    = idx & 15;
    int lane = (idx >> 4) & 31;
    int t    = idx >> 9;                 // kt*(N/16)+nt
    int ntl  = N >> 4;
    int kt = t / ntl, nt = t - kt * ntl;
    int kk = (e < 8 ? e : e + 8) + ((lane & 16) ? 8 : 0);
    int k  = kt * 32 + kk;
    int n  = nt * 16 + (lane & 15);
    P[idx] = (_Float16)W[k * N + n];
}

// ---------------- edge kernel ----------------
#define EIN_STRIDE 392                       // halves (384 + 8 pad)
#define EH_STRIDE  136
#define H2_STRIDE  133
#define EIN_BYTES  (128 * EIN_STRIDE * 2)    // 100352
#define EH_BYTES   (128 * EH_STRIDE * 2)     // 34816

__global__ __launch_bounds__(256)
void edge_kernel(const float* __restrict__ xn, const float* __restrict__ xh,
                 const float* __restrict__ ef,
                 const int* __restrict__ snd, const int* __restrict__ rcv,
                 const _Float16* __restrict__ w0p, const float* __restrict__ b0,
                 const _Float16* __restrict__ w1p, const float* __restrict__ b1,
                 const _Float16* __restrict__ w2p, const float* __restrict__ b2,
                 const float* __restrict__ g, const float* __restrict__ beta,
                 float* __restrict__ outE,
                 float* __restrict__ segsum, float* __restrict__ cnt,
                 unsigned* __restrict__ maxe, unsigned* __restrict__ mine) {
    __shared__ __align__(16) char smem[EIN_BYTES + EH_BYTES];
    __shared__ int sidx[128], ridx[128];
    _Float16* ein = (_Float16*)smem;
    _Float16* hb  = (_Float16*)(smem + EIN_BYTES);
    float*    h2  = (float*)smem;            // reuses ein region (L0 done by then)

    const int tid = threadIdx.x;
    const int e0  = blockIdx.x * 128;

    if (tid < 128) {
        int e = e0 + tid, s = 0, r = 0;
        if (e < NE) { s = snd[e]; r = rcv[e]; }
        sidx[tid] = s; ridx[tid] = r;
    }
    __syncthreads();

    // gather [sender | receiver | edge] -> LDS f16, 128 rows x 96 float4
#pragma unroll 4
    for (int j = 0; j < 48; ++j) {
        int idx = tid + j * 256;
        int row = idx / 96;
        int q   = idx - row * 96;
        float4 v = make_float4(0.f, 0.f, 0.f, 0.f);
        int e = e0 + row;
        if (e < NE) {
            if (q < 32) {
                int s = sidx[row];
                const float* src = (s < N_REG) ? (xn + (size_t)s * D)
                                               : (xh + (size_t)(s - N_REG) * D);
                v = *(const float4*)(src + q * 4);
            } else if (q < 64) {
                int r = ridx[row];
                const float* src = (r < N_REG) ? (xn + (size_t)r * D)
                                               : (xh + (size_t)(r - N_REG) * D);
                v = *(const float4*)(src + (q - 32) * 4);
            } else {
                v = *(const float4*)(ef + (size_t)e * D + (q - 64) * 4);
            }
        }
        v4h hv;
        hv[0] = (_Float16)v.x; hv[1] = (_Float16)v.y;
        hv[2] = (_Float16)v.z; hv[3] = (_Float16)v.w;
        *(v4h*)(ein + row * EIN_STRIDE + q * 4) = hv;
    }
    __syncthreads();

    const int lane = tid & 31;
    const int m0   = (tid >> 5) * 16;
    v8f acc[8];

    // layer 0: [128,384]@[384,128]
#pragma unroll
    for (int n = 0; n < 8; ++n) acc[n] = (v8f)0.f;
    gemm8<12>(ein, EIN_STRIDE, m0, w0p, lane, acc);
    __syncthreads();
    store_relu_f16(hb, EH_STRIDE, m0, lane, acc, b0);
    __syncthreads();

    // layer 1
#pragma unroll
    for (int n = 0; n < 8; ++n) acc[n] = (v8f)0.f;
    gemm8<4>(hb, EH_STRIDE, m0, w1p, lane, acc);
    __syncthreads();                         // all reads of hb done
    store_relu_f16(hb, EH_STRIDE, m0, lane, acc, b1);
    __syncthreads();

    // layer 2 (linear) -> fp32 staging
#pragma unroll
    for (int n = 0; n < 8; ++n) acc[n] = (v8f)0.f;
    gemm8<4>(hb, EH_STRIDE, m0, w2p, lane, acc);
    __syncthreads();
    store_f32(h2, H2_STRIDE, m0, lane, acc, b2);
    __syncthreads();

    // LayerNorm + residual + PNA scatter (2 threads per edge row)
    int r    = tid & 127;
    int half = tid >> 7;
    int e    = e0 + r;
    if (e < NE) {
        const float* row = h2 + r * H2_STRIDE;
        float s = 0.f, s2 = 0.f;
#pragma unroll 8
        for (int c = 0; c < D; ++c) { float v = row[c]; s += v; s2 += v * v; }
        float mean = s * (1.f / D);
        float var  = s2 * (1.f / D) - mean * mean;
        float rstd = rsqrtf(var + 1e-5f);
        int recv = ridx[r];
        float*    srow = segsum + (size_t)recv * D;
        unsigned* mxr  = maxe + (size_t)recv * D;
        unsigned* mnr  = mine + (size_t)recv * D;
        int cbeg = half * 64;
        for (int c = cbeg; c < cbeg + 64; ++c) {
            float v = (row[c] - mean) * rstd * g[c] + beta[c];
            outE[(size_t)e * D + c] = v + ef[(size_t)e * D + c];
            atomicAdd(&srow[c], v);
            atomicMax(&mxr[c], fenc(v));
            atomicMin(&mnr[c], fenc(v));
        }
        if (half == 0) atomicAdd(&cnt[recv], 1.0f);
    }
}

// ---------------- node kernel ----------------
#define FE_STRIDE 648                        // halves (640 + 8 pad)
#define FE_BYTES  (128 * FE_STRIDE * 2)      // 165888

__global__ __launch_bounds__(256)
void node_kernel(const float* __restrict__ xn,
                 const float* __restrict__ segsum, const float* __restrict__ cnt,
                 const unsigned* __restrict__ maxe, const unsigned* __restrict__ mine,
                 const _Float16* __restrict__ w0p, const float* __restrict__ b0,
                 const _Float16* __restrict__ w1p, const float* __restrict__ b1,
                 const _Float16* __restrict__ w2p, const float* __restrict__ b2,
                 const float* __restrict__ g, const float* __restrict__ beta,
                 float* __restrict__ outN) {
    __shared__ __align__(16) char smem[FE_BYTES + EH_BYTES];
    __shared__ float cntv[128];
    _Float16* fe = (_Float16*)smem;
    _Float16* hb = (_Float16*)(smem + FE_BYTES);
    float*    h2 = (float*)smem;

    const int tid = threadIdx.x;
    const int n0  = blockIdx.x * 128;

    if (tid < 128) {
        int n = n0 + tid;
        cntv[tid] = (n < N_REG) ? cnt[n] : 0.f;
    }
    __syncthreads();

    // gather feats = [x | sum | mean | max | min] -> 128 rows x 160 float4
#pragma unroll 4
    for (int j = 0; j < 80; ++j) {
        int idx = tid + j * 256;
        int row = idx / 160;
        int q   = idx - row * 160;
        int n   = n0 + row;
        float4 v = make_float4(0.f, 0.f, 0.f, 0.f);
        if (n < N_REG) {
            float c = cntv[row];
            if (q < 32) {
                v = *(const float4*)(xn + (size_t)n * D + q * 4);
            } else if (q < 64) {
                v = *(const float4*)(segsum + (size_t)n * D + (q - 32) * 4);
            } else if (q < 96) {
                float4 s = *(const float4*)(segsum + (size_t)n * D + (q - 64) * 4);
                float inv = (c > 0.f) ? (1.f / c) : 0.f;
                v = make_float4(s.x * inv, s.y * inv, s.z * inv, s.w * inv);
            } else if (q < 128) {
                uint4 m = *(const uint4*)(maxe + (size_t)n * D + (q - 96) * 4);
                if (c > 0.f) v = make_float4(fdec(m.x), fdec(m.y), fdec(m.z), fdec(m.w));
            } else {
                uint4 m = *(const uint4*)(mine + (size_t)n * D + (q - 128) * 4);
                if (c > 0.f) v = make_float4(fdec(m.x), fdec(m.y), fdec(m.z), fdec(m.w));
            }
        }
        v4h hv;
        hv[0] = (_Float16)v.x; hv[1] = (_Float16)v.y;
        hv[2] = (_Float16)v.z; hv[3] = (_Float16)v.w;
        *(v4h*)(fe + row * FE_STRIDE + q * 4) = hv;
    }
    __syncthreads();

    const int lane = tid & 31;
    const int m0   = (tid >> 5) * 16;
    v8f acc[8];

#pragma unroll
    for (int n = 0; n < 8; ++n) acc[n] = (v8f)0.f;
    gemm8<20>(fe, FE_STRIDE, m0, w0p, lane, acc);
    __syncthreads();
    store_relu_f16(hb, EH_STRIDE, m0, lane, acc, b0);
    __syncthreads();

#pragma unroll
    for (int n = 0; n < 8; ++n) acc[n] = (v8f)0.f;
    gemm8<4>(hb, EH_STRIDE, m0, w1p, lane, acc);
    __syncthreads();
    store_relu_f16(hb, EH_STRIDE, m0, lane, acc, b1);
    __syncthreads();

#pragma unroll
    for (int n = 0; n < 8; ++n) acc[n] = (v8f)0.f;
    gemm8<4>(hb, EH_STRIDE, m0, w2p, lane, acc);
    __syncthreads();
    store_f32(h2, H2_STRIDE, m0, lane, acc, b2);
    __syncthreads();

    int r    = tid & 127;
    int half = tid >> 7;
    int n    = n0 + r;
    if (n < N_REG) {
        const float* row = h2 + r * H2_STRIDE;
        float s = 0.f, s2 = 0.f;
#pragma unroll 8
        for (int c = 0; c < D; ++c) { float v = row[c]; s += v; s2 += v * v; }
        float mean = s * (1.f / D);
        float var  = s2 * (1.f / D) - mean * mean;
        float rstd = rsqrtf(var + 1e-5f);
        int cbeg = half * 64;
        for (int c = cbeg; c < cbeg + 64; ++c) {
            float v = (row[c] - mean) * rstd * g[c] + beta[c];
            outN[(size_t)n * D + c] = v + xn[(size_t)n * D + c];
        }
    }
}

// ---------------- hyper kernel ----------------
__global__ void hyper_kernel(const float* __restrict__ xh, float* __restrict__ outH) {
    int i = blockIdx.x * 256 + threadIdx.x;
    if (i < N_HYP * D) outH[i] = 2.0f * xh[i];
}

// ---------------- launch ----------------
extern "C" void kernel_launch(void* const* d_in, const int* in_sizes, int n_in,
                              void* d_out, int out_size, void* d_ws, size_t ws_size,
                              hipStream_t stream) {
    (void)in_sizes; (void)n_in; (void)out_size;
    if (ws_size < WS_NEED) return;

    const float* xn   = (const float*)d_in[0];
    const float* xh   = (const float*)d_in[1];
    const float* ef   = (const float*)d_in[2];
    const int*   snd  = (const int*)d_in[3];
    const int*   rcv  = (const int*)d_in[4];
    const float* ew0  = (const float*)d_in[5];
    const float* eb0  = (const float*)d_in[6];
    const float* ew1  = (const float*)d_in[7];
    const float* eb1  = (const float*)d_in[8];
    const float* ew2  = (const float*)d_in[9];
    const float* eb2  = (const float*)d_in[10];
    const float* eg   = (const float*)d_in[11];
    const float* ebt  = (const float*)d_in[12];
    const float* nw0  = (const float*)d_in[13];
    const float* nb0  = (const float*)d_in[14];
    const float* nw1  = (const float*)d_in[15];
    const float* nb1  = (const float*)d_in[16];
    const float* nw2  = (const float*)d_in[17];
    const float* nb2  = (const float*)d_in[18];
    const float* ng   = (const float*)d_in[19];
    const float* nbt  = (const float*)d_in[20];

    char* ws = (char*)d_ws;
    _Float16* ew0p = (_Float16*)(ws + WS_EW0P);
    _Float16* ew1p = (_Float16*)(ws + WS_EW1P);
    _Float16* ew2p = (_Float16*)(ws + WS_EW2P);
    _Float16* nw0p = (_Float16*)(ws + WS_NW0P);
    _Float16* nw1p = (_Float16*)(ws + WS_NW1P);
    _Float16* nw2p = (_Float16*)(ws + WS_NW2P);
    float*    segs = (float*)(ws + WS_SUM);
    float*    cnt  = (float*)(ws + WS_CNT);
    unsigned* maxe = (unsigned*)(ws + WS_MAXE);
    unsigned* mine = (unsigned*)(ws + WS_MINE);

    float* outN = (float*)d_out;
    float* outH = outN + (size_t)N_REG * D;
    float* outE = outN + (size_t)N_TOT * D;

    // init aggregation buffers
    init_ws_kernel<<<4096, 256, 0, stream>>>(segs, cnt, maxe, mine);

    // pack weights into WMMA B-fragment order (f16)
    pack_weights_kernel<<<(384 * 128 + 255) / 256, 256, 0, stream>>>(ew0, ew0p, 384, 128);
    pack_weights_kernel<<<(128 * 128 + 255) / 256, 256, 0, stream>>>(ew1, ew1p, 128, 128);
    pack_weights_kernel<<<(128 * 128 + 255) / 256, 256, 0, stream>>>(ew2, ew2p, 128, 128);
    pack_weights_kernel<<<(640 * 128 + 255) / 256, 256, 0, stream>>>(nw0, nw0p, 640, 128);
    pack_weights_kernel<<<(128 * 128 + 255) / 256, 256, 0, stream>>>(nw1, nw1p, 128, 128);
    pack_weights_kernel<<<(128 * 128 + 255) / 256, 256, 0, stream>>>(nw2, nw2p, 128, 128);

    // edge MLP + fused PNA scatter + edge residual
    edge_kernel<<<(NE + 127) / 128, 256, 0, stream>>>(
        xn, xh, ef, snd, rcv, ew0p, eb0, ew1p, eb1, ew2p, eb2, eg, ebt,
        outE, segs, cnt, maxe, mine);

    // node MLP + residual
    node_kernel<<<(N_REG + 127) / 128, 256, 0, stream>>>(
        xn, segs, cnt, maxe, mine, nw0p, nb0, nw1p, nb1, nw2p, nb2, ng, nbt, outN);

    // hyper residual
    hyper_kernel<<<(N_HYP * D + 255) / 256, 256, 0, stream>>>(xh, outH);
}
